// SemanticOrdering_1460288881207
// MI455X (gfx1250) — matrix operand order repaired and checked by reading, not compiled, not run on hardware
//
#include <hip/hip_runtime.h>
#include <hip/hip_bf16.h>
#include <math.h>

// Problem constants (match reference)
#define B_DIM 16
#define N_DIM 2048
#define D_DIM 768
#define LAMBDA_SPATIAL 0.5f
#define INV_SIGMA2 (1.0f / (100.0f * 100.0f))

typedef __attribute__((ext_vector_type(16))) __bf16 v16bf;
typedef __attribute__((ext_vector_type(8)))  float  v8f;

// ---------------- helpers ----------------

__device__ __forceinline__ unsigned short f32_to_bf16_rne(float f) {
    unsigned int u = __float_as_uint(f);
    unsigned int rnd = 0x7FFFu + ((u >> 16) & 1u);
    u += rnd;
    return (unsigned short)(u >> 16);
}

__device__ __forceinline__ void pair_max(float& v, int& i, float ov, int oi) {
    // prefer larger value; on tie prefer LOWER index (matches jnp.argmax)
    if (ov > v || (ov == v && oi < i)) { v = ov; i = oi; }
}

__device__ __forceinline__ void wave_argmax(float& v, int& i) {
#pragma unroll
    for (int off = 16; off > 0; off >>= 1) {
        float ov = __shfl_xor(v, off, 32);
        int   oi = __shfl_xor(i, off, 32);
        pair_max(v, i, ov, oi);
    }
}

// ---------------- kernel 1: normalize features, convert to bf16 ----------------
// one 256-thread block per row (B*N rows)
__global__ __launch_bounds__(256) void normalize_kernel(const float* __restrict__ feat,
                                                        unsigned short* __restrict__ fn) {
    const int row = blockIdx.x;                 // 0 .. B*N-1
    const float* src = feat + (size_t)row * D_DIM;
    unsigned short* dst = fn + (size_t)row * D_DIM;

    float s = 0.0f;
    for (int d = threadIdx.x; d < D_DIM; d += 256) {
        float v = src[d];
        s += v * v;
    }
#pragma unroll
    for (int off = 16; off > 0; off >>= 1) s += __shfl_xor(s, off, 32);

    __shared__ float red[8];
    if ((threadIdx.x & 31) == 0) red[threadIdx.x >> 5] = s;
    __syncthreads();
    float tot = 0.0f;
#pragma unroll
    for (int w = 0; w < 8; ++w) tot += red[w];
    float nrm = sqrtf(tot);
    float inv = 1.0f / fmaxf(nrm, 1e-12f);

    for (int d = threadIdx.x; d < D_DIM; d += 256) {
        dst[d] = f32_to_bf16_rne(src[d] * inv);
    }
}

// ---------------- kernel 2: WMMA GEMM S = fn*fn^T + lambda*exp(-d^2/sigma^2) ----------------
// grid: (N/16/8, N/64, B), block: 256 threads = 8 waves.
// Each wave computes a 16(M) x 64(N) strip: 4 accumulators, A fragment reused 4x.
__global__ __launch_bounds__(256) void gemm_sim_kernel(const unsigned short* __restrict__ fnraw,
                                                       const float* __restrict__ coords,
                                                       float* __restrict__ S) {
    const __bf16* fn = (const __bf16*)fnraw;
    const int wave = threadIdx.x >> 5;
    const int lane = threadIdx.x & 31;
    const int m    = lane & 15;           // row/col within 16-wide tile
    const int hi   = lane >> 4;           // 0 for lanes 0-15, 1 for lanes 16-31

    const int b       = blockIdx.z;
    const int rowBase = (blockIdx.x * 8 + wave) * 16;   // M tile base
    const int colBase = blockIdx.y * 64;                // N strip base

    const __bf16* A  = fn + ((size_t)b * N_DIM + rowBase) * D_DIM;
    const __bf16* Bm = fn + ((size_t)b * N_DIM + colBase) * D_DIM;

    v8f acc[4] = {v8f{}, v8f{}, v8f{}, v8f{}};

    // A-fragment row pointer for this lane (ISA 16-bit A 16x32 layout):
    //   lanes 0-15 hold K = [0..7] and [16..23]; lanes 16-31 hold K = [8..15] and [24..31]
    const __bf16* arow = A + (size_t)m * D_DIM + hi * 8;

    for (int k = 0; k < D_DIM; k += 32) {
        v16bf a;
#pragma unroll
        for (int i = 0; i < 8; ++i) {
            a[i]     = arow[k + i];
            a[8 + i] = arow[k + 16 + i];
        }
#pragma unroll
        for (int t = 0; t < 4; ++t) {
            // B-fragment (32x16): same K order as A: lanes 0-15 hold K=[0..7],[16..23]; lanes 16-31 hold K=[8..15],[24..31],
            // column = lane%16 -> row (colBase + t*16 + m)
            const __bf16* brow = Bm + (size_t)(t * 16 + m) * D_DIM + k + hi * 8;
            v16bf bb;
#pragma unroll
            for (int j = 0; j < 8; ++j) { bb[j] = brow[j]; bb[8 + j] = brow[16 + j]; }
            acc[t] = __builtin_amdgcn_wmma_f32_16x16x32_bf16(
                false, a, false, bb, (short)0, acc[t], false, false);
        }
    }

    // Epilogue: add spatial Gaussian term and store.
    const float* cb = coords + (size_t)b * N_DIM * 2;
    float jx[4], jy[4];
#pragma unroll
    for (int t = 0; t < 4; ++t) {
        int j = colBase + t * 16 + m;
        jx[t] = cb[2 * j];
        jy[t] = cb[2 * j + 1];
    }
    float* Sb = S + (size_t)b * N_DIM * N_DIM;
#pragma unroll
    for (int r = 0; r < 8; ++r) {
        int i = rowBase + r + hi * 8;     // C/D layout: VGPR r -> M = r (+8 for lanes 16-31)
        float ix = cb[2 * i], iy = cb[2 * i + 1];
        float* srow = Sb + (size_t)i * N_DIM;
#pragma unroll
        for (int t = 0; t < 4; ++t) {
            float dx = ix - jx[t];
            float dy = iy - jy[t];
            float sp = __expf(-(dx * dx + dy * dy) * INV_SIGMA2);
            srow[colBase + t * 16 + m] = acc[t][r] + LAMBDA_SPATIAL * sp;
        }
    }
}

// ---------------- kernel 3: row sums of S (for start-node argmax) ----------------
// one 256-thread block per row (B*N rows)
__global__ __launch_bounds__(256) void rowsum_kernel(const float* __restrict__ S,
                                                     float* __restrict__ rowsum) {
    const int row = blockIdx.x;                 // 0 .. B*N-1
    const float* src = S + (size_t)row * N_DIM;
    float s = 0.0f;
    for (int j = threadIdx.x; j < N_DIM; j += 256) s += src[j];
#pragma unroll
    for (int off = 16; off > 0; off >>= 1) s += __shfl_xor(s, off, 32);
    __shared__ float red[8];
    if ((threadIdx.x & 31) == 0) red[threadIdx.x >> 5] = s;
    __syncthreads();
    if (threadIdx.x == 0) {
        float tot = 0.0f;
#pragma unroll
        for (int w = 0; w < 8; ++w) tot += red[w];
        rowsum[row] = tot;
    }
}

// ---------------- kernel 4: sequential greedy nearest-neighbor order ----------------
// one 1024-thread block per batch (16 blocks); 2048 dependent argmax steps
__global__ __launch_bounds__(1024) void greedy_kernel(const float* __restrict__ S,
                                                      const float* __restrict__ rowsum,
                                                      int* __restrict__ order) {
    const int b = blockIdx.x;
    const int t = threadIdx.x;
    const int lane = t & 31;
    const int wv   = t >> 5;                    // 0..31

    __shared__ unsigned char visited[N_DIM];
    __shared__ float wval[32];
    __shared__ int   widx[32];
    __shared__ int   curSh;

    for (int j = t; j < N_DIM; j += 1024) visited[j] = 0;
    __syncthreads();

    const float* Sb = S + (size_t)b * N_DIM * N_DIM;
    const float* rs = rowsum + (size_t)b * N_DIM;
    int* ob = order + (size_t)b * N_DIM;

    // ---- start = argmax(rowsum) ----
    {
        float v0 = rs[t], v1 = rs[t + 1024];
        float bv = v0; int bi = t;
        pair_max(bv, bi, v1, t + 1024);
        wave_argmax(bv, bi);
        if (lane == 0) { wval[wv] = bv; widx[wv] = bi; }
        __syncthreads();
        if (wv == 0) {
            float v = wval[lane]; int i = widx[lane];
            wave_argmax(v, i);
            if (lane == 0) {
                curSh = i;
                visited[i] = 1;
                ob[0] = i;
            }
        }
    }

    // ---- 2047 dependent steps ----
    for (int step = 1; step < N_DIM; ++step) {
        __syncthreads();                        // publish curSh / visited
        const int cur = curSh;
        const float* row = Sb + (size_t)cur * N_DIM;

        float v0 = visited[t]        ? -INFINITY : row[t];
        float v1 = visited[t + 1024] ? -INFINITY : row[t + 1024];
        float bv = v0; int bi = t;
        pair_max(bv, bi, v1, t + 1024);
        wave_argmax(bv, bi);
        if (lane == 0) { wval[wv] = bv; widx[wv] = bi; }
        __syncthreads();
        if (wv == 0) {
            float v = wval[lane]; int i = widx[lane];
            wave_argmax(v, i);
            if (lane == 0) {
                curSh = i;
                visited[i] = 1;
                ob[step] = i;
            }
        }
    }
}

// ---------------- kernel 5: gather reorder + emit order ----------------
// one 256-thread block per output row (B*N rows)
__global__ __launch_bounds__(256) void reorder_kernel(const float* __restrict__ feat,
                                                      const int* __restrict__ order,
                                                      float* __restrict__ outFeat,
                                                      float* __restrict__ outOrder) {
    const int row = blockIdx.x;                 // b*N + i
    const int b = row / N_DIM;
    const int src = order[row];
    const float* in = feat + ((size_t)b * N_DIM + src) * D_DIM;
    float* out = outFeat + (size_t)row * D_DIM;
    for (int d = threadIdx.x; d < D_DIM; d += 256) out[d] = in[d];
    if (threadIdx.x == 0) outOrder[row] = (float)src;
}

// ---------------- launch ----------------
extern "C" void kernel_launch(void* const* d_in, const int* in_sizes, int n_in,
                              void* d_out, int out_size, void* d_ws, size_t ws_size,
                              hipStream_t stream) {
    const float* features = (const float*)d_in[0];   // [B,N,D] f32
    const float* coords   = (const float*)d_in[1];   // [B,N,2] f32

    // workspace layout
    const size_t FN_BYTES = (size_t)B_DIM * N_DIM * D_DIM * sizeof(unsigned short); // 48 MB
    const size_t S_BYTES  = (size_t)B_DIM * N_DIM * N_DIM * sizeof(float);          // 256 MB
    const size_t RS_BYTES = (size_t)B_DIM * N_DIM * sizeof(float);                  // 128 KB

    unsigned short* fn   = (unsigned short*)d_ws;
    float* S             = (float*)((char*)d_ws + FN_BYTES);
    float* rowsum        = (float*)((char*)d_ws + FN_BYTES + S_BYTES);
    int*   order         = (int*)  ((char*)d_ws + FN_BYTES + S_BYTES + RS_BYTES);

    float* outFeat  = (float*)d_out;                                   // B*N*D
    float* outOrder = (float*)d_out + (size_t)B_DIM * N_DIM * D_DIM;   // B*N

    // 1) normalize -> bf16
    normalize_kernel<<<B_DIM * N_DIM, 256, 0, stream>>>(features, fn);

    // 2) similarity GEMM via WMMA (each wave: 16x64 strip; block: 128 rows x 64 cols)
    dim3 ggrid(N_DIM / 16 / 8, N_DIM / 64, B_DIM);  // (16, 32, 16)
    gemm_sim_kernel<<<ggrid, 256, 0, stream>>>(fn, coords, S);

    // 3) row sums for start selection
    rowsum_kernel<<<B_DIM * N_DIM, 256, 0, stream>>>(S, rowsum);

    // 4) greedy traversal (one block per batch)
    greedy_kernel<<<B_DIM, 1024, 0, stream>>>(S, rowsum, order);

    // 5) gather + emit order
    reorder_kernel<<<B_DIM * N_DIM, 256, 0, stream>>>(features, order, outFeat, outOrder);
}
